// TransformerDecoderLayer_90855738180372
// MI455X (gfx1250) — compile-verified, matrix operand order checked by reading, not run on hardware
//
#include <hip/hip_runtime.h>

#define BM 128
#define BN 128
#define BK 64   // K-depth per staged tile (two 32-deep WMMA sub-steps)
#define KP 72   // padded LDS K-stride (elements); 72*2=144 bytes, multiple of 16

typedef unsigned short u16;
typedef __attribute__((ext_vector_type(16))) __bf16 v16bf;
typedef __attribute__((ext_vector_type(8)))  __bf16 v8bf;
typedef __attribute__((ext_vector_type(8)))  float  v8f;

#if defined(__HIP_DEVICE_COMPILE__) && \
    __has_builtin(__builtin_amdgcn_global_load_async_to_lds_b128)
#define HAVE_ASYNC_LDS 1
#else
#define HAVE_ASYNC_LDS 0
#endif

#if HAVE_ASYNC_LDS
typedef int v4i_t __attribute__((vector_size(16)));
typedef __attribute__((address_space(1))) v4i_t* as1v4;  // global int4*
typedef __attribute__((address_space(3))) v4i_t* as3v4;  // LDS int4*
__device__ __forceinline__ void asyncCp128(const u16* g, u16* l) {
  // global -> LDS, 16B per lane, tracked by ASYNCcnt (no VGPR round-trip)
  __builtin_amdgcn_global_load_async_to_lds_b128(
      (as1v4)(unsigned long long)g, (as3v4)(unsigned long long)l, 0, 0);
}
__device__ __forceinline__ void waitAsync() {
#if __has_builtin(__builtin_amdgcn_s_wait_asynccnt)
  __builtin_amdgcn_s_wait_asynccnt(0);
#else
  asm volatile("s_wait_asynccnt 0x0" ::: "memory");
#endif
}
#endif

__device__ __forceinline__ u16 f2bf(float f) {
  unsigned int u = __float_as_uint(f);
  u += 0x7FFFu + ((u >> 16) & 1u);   // round-to-nearest-even
  return (u16)(u >> 16);
}

struct GemmP {
  const u16* A; long long lda;       // A: [M,K] bf16 row-major
  const u16* B; long long ldb;       // Bt: [N,K] bf16 row-major (= B transposed)
  float* outF; long long ldcF;       // optional fp32 output
  u16*   outH; long long ldcH;       // optional bf16 output
  const float* bias;                 // optional, indexed by n
  const float* alphaPtr;             // optional device scalar
  const float* resid; long long ldr; // optional fp32 residual [M,N]
  int M, N, K, act;                  // act: 0 none, 1 relu, 2 leaky(0.01)
  int batchH;                        // z -> (b = z/batchH, h = z%batchH)
  long long aSB, aSH, bSB, bSH, cSB, cSH;
};

__global__ __launch_bounds__(256)
void k_gemm_bf16(GemmP p) {
  __shared__ u16 shA[2][BM * KP];    // 2 x 18432 B
  __shared__ u16 shB[2][BN * KP];    // 2 x 18432 B  (total 73728 B of 320KB WGP LDS)

  const int tid  = threadIdx.x;
  const int lane = tid & 31;
  const int wid  = tid >> 5;
  const int wm   = wid & 3;    // 4 wave rows of 32
  const int wn   = wid >> 2;   // 2 wave cols of 64
  const int l16  = lane & 15;
  const int hc   = lane >> 4;  // half-wave selector

  const int z  = blockIdx.z;
  const int bb = z / p.batchH;
  const int hh = z % p.batchH;
  const u16* A = p.A + (long long)bb * p.aSB + (long long)hh * p.aSH;
  const u16* B = p.B + (long long)bb * p.bSB + (long long)hh * p.bSH;
  const long long coff = (long long)bb * p.cSB + (long long)hh * p.cSH;

  const int m0 = blockIdx.y * BM;
  const int n0 = blockIdx.x * BN;

  // ---- loop-invariant staging descriptors: 4 x 16B chunks per matrix/thread ----
  const u16* gA[4];
  const u16* gB[4];
  int  sOf[4];
  bool bok[4];
#pragma unroll
  for (int u = 0; u < 4; ++u) {
    const int idx = tid + u * 256;           // 1024 chunks over 128 rows x 64 cols
    const int r = idx >> 3, c = (idx & 7) * 8;
    gA[u] = A + (long long)(m0 + r) * p.lda + c;
    gB[u] = B + (long long)(n0 + r) * p.ldb + c;
    sOf[u] = r * KP + c;
    bok[u] = (n0 + r) < p.N;
  }

  v8f acc[2][4];
  v8f z8 = {0.f, 0.f, 0.f, 0.f, 0.f, 0.f, 0.f, 0.f};
#pragma unroll
  for (int i = 0; i < 2; ++i)
#pragma unroll
    for (int j = 0; j < 4; ++j) acc[i][j] = z8;

  const int nk = p.K / BK;

#if HAVE_ASYNC_LDS
  // prologue: stage tile 0 into buffer 0
#pragma unroll
  for (int u = 0; u < 4; ++u) {
    asyncCp128(gA[u], &shA[0][sOf[u]]);
    if (bok[u]) asyncCp128(gB[u], &shB[0][sOf[u]]);
  }
  waitAsync();
  __syncthreads();
#endif

  for (int kt = 0; kt < nk; ++kt) {
#if HAVE_ASYNC_LDS
    const int cur = kt & 1;
    if (kt + 1 < nk) {  // prefetch next 128x64 tile into the other buffer
      const long long ko = (long long)(kt + 1) * BK;
      const int nxt = cur ^ 1;
#pragma unroll
      for (int u = 0; u < 4; ++u) {
        asyncCp128(gA[u] + ko, &shA[nxt][sOf[u]]);
        if (bok[u]) asyncCp128(gB[u] + ko, &shB[nxt][sOf[u]]);
      }
    }
#else
    const int cur = 0;
    const long long ko = (long long)kt * BK;
    __syncthreads();
#pragma unroll
    for (int u = 0; u < 4; ++u) {
      int4 av = *(const int4*)(gA[u] + ko);
      int4 bv = {0, 0, 0, 0};
      if (bok[u]) bv = *(const int4*)(gB[u] + ko);
      *(int4*)(&shA[0][sOf[u]]) = av;
      *(int4*)(&shB[0][sOf[u]]) = bv;
    }
    if (kt + 1 < nk) {
      __builtin_prefetch((const void*)(gA[0] + ko + BK), 0, 1);
      __builtin_prefetch((const void*)(gB[0] + ko + BK), 0, 1);
    }
    __syncthreads();
#endif

    // --- two 32-deep sub-steps; fragments per ISA 16-bit A/B layout ---
#pragma unroll
    for (int s = 0; s < 2; ++s) {
      const int kb = s * 32;
      union V16 { v16bf v; v8bf h[2]; };
      V16 afr[2], bfr[4];
#pragma unroll
      for (int i = 0; i < 2; ++i) {
        const int r = wm * 32 + i * 16 + l16;
        afr[i].h[0] = *(const v8bf*)&shA[cur][r * KP + kb + 8 * hc];
        afr[i].h[1] = *(const v8bf*)&shA[cur][r * KP + kb + 16 + 8 * hc];
      }
#pragma unroll
      for (int j = 0; j < 4; ++j) {
        const int r = wn * 64 + j * 16 + l16;
        bfr[j].h[0] = *(const v8bf*)&shB[cur][r * KP + kb + 8 * hc];
        bfr[j].h[1] = *(const v8bf*)&shB[cur][r * KP + kb + 16 + 8 * hc];
      }
#pragma unroll
      for (int i = 0; i < 2; ++i)
#pragma unroll
        for (int j = 0; j < 4; ++j)
          acc[i][j] = __builtin_amdgcn_wmma_f32_16x16x32_bf16(
              false, afr[i].v, false, bfr[j].v, (short)0, acc[i][j], false, false);
    }

#if HAVE_ASYNC_LDS
    if (kt + 1 < nk) {
      waitAsync();      // prefetched tile landed in LDS
      __syncthreads();  // all waves done reading current buffer
    }
#endif
  }

  // --- fused epilogue ---
  const float alpha = p.alphaPtr ? p.alphaPtr[0] : 1.0f;
#pragma unroll
  for (int i = 0; i < 2; ++i) {
#pragma unroll
    for (int j = 0; j < 4; ++j) {
      const int col = n0 + wn * 64 + j * 16 + l16;
      if (col >= p.N) continue;
      const int rowB = m0 + wm * 32 + i * 16 + hc * 8;
      const float bsv = p.bias ? p.bias[col] : 0.0f;
#pragma unroll
      for (int r = 0; r < 8; ++r) {
        const int row = rowB + r;
        float v = acc[i][j][r] + bsv;
        if (p.act == 1)      v = fmaxf(v, 0.0f);
        else if (p.act == 2) v = (v > 0.0f) ? v : 0.01f * v;
        v *= alpha;
        if (p.resid) v += p.resid[(long long)row * p.ldr + col];
        if (p.outF) p.outF[coff + (long long)row * p.ldcF + col] = v;
        if (p.outH) p.outH[coff + (long long)row * p.ldcH + col] = f2bf(v);
      }
    }
  }
}

// fp32 -> bf16, optional destination row stride (for the concat buffer)
__global__ void k_cvt(const float* __restrict__ s, u16* __restrict__ d,
                      long long n, int cols, int dstLd) {
  long long i = (long long)blockIdx.x * blockDim.x + threadIdx.x;
  if (i >= n) return;
  long long r = i / cols, c = i % cols;
  d[r * dstLd + c] = f2bf(s[i]);
}

// fp32 [R,C] -> bf16 [C,R]  (weight / embedding transpose-convert)
__global__ void k_cvt_t(const float* __restrict__ s, u16* __restrict__ d, int R, int C) {
  __shared__ float t[32][33];
  const int c0 = blockIdx.x * 32, r0 = blockIdx.y * 32;
  for (int i = threadIdx.y; i < 32; i += 8)
    t[i][threadIdx.x] = s[(long long)(r0 + i) * C + c0 + threadIdx.x];
  __syncthreads();
  for (int i = threadIdx.y; i < 32; i += 8)
    d[(long long)(c0 + i) * R + r0 + threadIdx.x] = f2bf(t[threadIdx.x][i]);
}

// bf16 batched transpose: per z, src[R,C] (ld=srcLd) -> dst[C,R] (ld=R)
__global__ void k_tr_bf16(const u16* __restrict__ s, u16* __restrict__ d,
                          int R, int C, long long srcLd, int batchH,
                          long long sSB, long long sSH, long long dS) {
  const int z = blockIdx.z;
  const int bb = z / batchH, hh = z % batchH;
  const u16* sp = s + (long long)bb * sSB + (long long)hh * sSH;
  u16* dp = d + (long long)z * dS;
  __shared__ u16 t[32][33];
  const int r0 = blockIdx.y * 32, c0 = blockIdx.x * 32;
  for (int i = threadIdx.y; i < 32; i += 8)
    t[i][threadIdx.x] = sp[(long long)(r0 + i) * srcLd + c0 + threadIdx.x];
  __syncthreads();
  for (int i = threadIdx.y; i < 32; i += 8)
    dp[(long long)(c0 + i) * R + r0 + threadIdx.x] = t[threadIdx.x][i];
}

// row softmax: fp32 in (scaled), bf16 out; one block per row
__global__ __launch_bounds__(256)
void k_softmax(const float* __restrict__ in, u16* __restrict__ out,
               int rowLen, float scale) {
  __shared__ float red[256];
  const long long row = blockIdx.x;
  const float* p = in + row * rowLen;
  u16* q = out + row * rowLen;
  const int tid = threadIdx.x;

  float m = -3.4e38f;
  for (int i = tid; i < rowLen; i += 256) m = fmaxf(m, p[i] * scale);
  red[tid] = m; __syncthreads();
  for (int s = 128; s > 0; s >>= 1) {
    if (tid < s) red[tid] = fmaxf(red[tid], red[tid + s]);
    __syncthreads();
  }
  m = red[0]; __syncthreads();

  float sum = 0.f;
  for (int i = tid; i < rowLen; i += 256) sum += __expf(p[i] * scale - m);
  red[tid] = sum; __syncthreads();
  for (int s = 128; s > 0; s >>= 1) {
    if (tid < s) red[tid] += red[tid + s];
    __syncthreads();
  }
  const float inv = 1.0f / red[0];
  for (int i = tid; i < rowLen; i += 256)
    q[i] = f2bf(__expf(p[i] * scale - m) * inv);
}

extern "C" void kernel_launch(void* const* d_in, const int* in_sizes, int n_in,
                              void* d_out, int out_size, void* d_ws, size_t ws_size,
                              hipStream_t stream) {
  (void)in_sizes; (void)n_in; (void)out_size; (void)ws_size;
  const int Bz = 4, L = 1024, D = 1024, DFF = 2048, V = 32000, H = 16, HD = 64;
  const int M4 = Bz * L;

  const float* tgt   = (const float*)d_in[0];
  const float* srcp  = (const float*)d_in[1];
  const float* emb   = (const float*)d_in[2];
  const float* va_w  = (const float*)d_in[3];
  const float* va_b  = (const float*)d_in[4];
  const float* a_va  = (const float*)d_in[5];
  const float* sa_wq = (const float*)d_in[6];
  const float* sa_bq = (const float*)d_in[7];
  const float* sa_wk = (const float*)d_in[8];
  const float* sa_bk = (const float*)d_in[9];
  const float* sa_wv = (const float*)d_in[10];
  const float* sa_bv = (const float*)d_in[11];
  const float* sa_wo = (const float*)d_in[12];
  const float* sa_bo = (const float*)d_in[13];
  const float* a_sa  = (const float*)d_in[14];
  const float* ca_wq = (const float*)d_in[15];
  const float* ca_bq = (const float*)d_in[16];
  const float* ca_wk = (const float*)d_in[17];
  const float* ca_bk = (const float*)d_in[18];
  const float* ca_wv = (const float*)d_in[19];
  const float* ca_bv = (const float*)d_in[20];
  const float* ca_wo = (const float*)d_in[21];
  const float* ca_bo = (const float*)d_in[22];
  const float* a_ca  = (const float*)d_in[23];
  const float* ff_w1 = (const float*)d_in[24];
  const float* ff_b1 = (const float*)d_in[25];
  const float* ff_w2 = (const float*)d_in[26];
  const float* ff_b2 = (const float*)d_in[27];
  const float* a_ff  = (const float*)d_in[28];

  char* ws = (char*)d_ws;
  size_t off = 0;
  auto take = [&](size_t bytes) -> void* {
    void* r = ws + off;
    off += (bytes + 255) & ~(size_t)255;
    return r;
  };

  u16* E_bf   = (u16*)take((size_t)V * D * 2);       // embedding bf16 [V,D]
  u16* Et_bf  = (u16*)take((size_t)V * D * 2);       // embedding^T bf16 [D,V]
  u16* vaWt   = (u16*)take((size_t)2 * D * D * 2);   // [D, 2D]
  u16* saWqT  = (u16*)take((size_t)D * D * 2);
  u16* saWkT  = (u16*)take((size_t)D * D * 2);
  u16* saWvT  = (u16*)take((size_t)D * D * 2);
  u16* saWoT  = (u16*)take((size_t)D * D * 2);
  u16* caWqT  = (u16*)take((size_t)D * D * 2);
  u16* caWkT  = (u16*)take((size_t)D * D * 2);
  u16* caWvT  = (u16*)take((size_t)D * D * 2);
  u16* caWoT  = (u16*)take((size_t)D * D * 2);
  u16* ff1T   = (u16*)take((size_t)D * DFF * 2);     // [DFF, D]
  u16* ff2T   = (u16*)take((size_t)DFF * D * 2);     // [D, DFF]
  float* x_f  = (float*)take((size_t)M4 * D * 4);    // fp32 residual stream
  u16* x_bf   = (u16*)take((size_t)M4 * D * 2);
  u16* src_bf = (u16*)take((size_t)M4 * D * 2);
  u16* xa_bf  = (u16*)take((size_t)M4 * 2 * D * 2);  // concat / ff hidden
  u16* q_bf   = (u16*)take((size_t)M4 * D * 2);
  u16* k_bf   = (u16*)take((size_t)M4 * D * 2);
  u16* v_bf   = (u16*)take((size_t)M4 * D * 2);
  u16* vt_bf  = (u16*)take((size_t)M4 * D * 2);      // [B*H, HD, L]
  u16* ctx_bf = (u16*)take((size_t)M4 * D * 2);
  float* logits = (float*)take((size_t)M4 * V * 4);  // reused for attn scores
  u16* probs    = (u16*)take((size_t)M4 * V * 2);    // reused for attn probs
  float* scores = logits;
  u16* aprobs   = probs;
  u16* hid      = xa_bf;

  auto cvt = [&](const float* s, u16* d, long long n, int cols, int ld) {
    k_cvt<<<(unsigned)((n + 255) / 256), 256, 0, stream>>>(s, d, n, cols, ld);
  };
  auto cvtT = [&](const float* s, u16* d, int R, int C) {
    k_cvt_t<<<dim3(C / 32, R / 32), dim3(32, 8), 0, stream>>>(s, d, R, C);
  };
  auto gemm = [&](const u16* A, long long lda, const u16* Bt, long long ldb,
                  float* oF, long long ldcF, u16* oH, long long ldcH,
                  const float* bias, const float* alphaPtr,
                  const float* resid, long long ldr,
                  int M, int N, int K, int act, int Z, int batchH,
                  long long aSB, long long aSH, long long bSB, long long bSH,
                  long long cSB, long long cSH) {
    GemmP p{A, lda, Bt, ldb, oF, ldcF, oH, ldcH, bias, alphaPtr, resid, ldr,
            M, N, K, act, batchH, aSB, aSH, bSB, bSH, cSB, cSH};
    dim3 g((N + BN - 1) / BN, M / BM, Z);
    k_gemm_bf16<<<g, dim3(256), 0, stream>>>(p);
  };

  // ---- precision prep: convert + transpose weights/activations to bf16 ----
  cvt(emb, E_bf, (long long)V * D, D, D);
  cvtT(emb, Et_bf, V, D);
  cvtT(va_w, vaWt, 2 * D, D);
  cvtT(sa_wq, saWqT, D, D); cvtT(sa_wk, saWkT, D, D);
  cvtT(sa_wv, saWvT, D, D); cvtT(sa_wo, saWoT, D, D);
  cvtT(ca_wq, caWqT, D, D); cvtT(ca_wk, caWkT, D, D);
  cvtT(ca_wv, caWvT, D, D); cvtT(ca_wo, caWoT, D, D);
  cvtT(ff_w1, ff1T, D, DFF);
  cvtT(ff_w2, ff2T, DFF, D);
  cvt(tgt, x_bf, (long long)M4 * D, D, D);
  cvt(tgt, xa_bf, (long long)M4 * D, D, 2 * D);  // concat first half = x
  cvt(srcp, src_bf, (long long)M4 * D, D, D);

  // ---- vocabulary attention: softmax(x @ E^T) @ E ----
  gemm(x_bf, D, E_bf, D, logits, V, nullptr, 0, nullptr, nullptr, nullptr, 0,
       M4, V, D, 0, 1, 1, 0, 0, 0, 0, 0, 0);
  k_softmax<<<M4, 256, 0, stream>>>(logits, probs, V, 1.0f);
  gemm(probs, V, Et_bf, V, nullptr, 0, xa_bf + D, 2 * D, nullptr, nullptr,
       nullptr, 0, M4, D, V, 0, 1, 1, 0, 0, 0, 0, 0, 0);
  // va = concat @ va_w + b ; x = tgt + alpha_va * va   (residual fused)
  gemm(xa_bf, 2 * D, vaWt, 2 * D, x_f, D, x_bf, D, va_b, a_va, tgt, D,
       M4, D, 2 * D, 0, 1, 1, 0, 0, 0, 0, 0, 0);

  // ---- multi-head attention (shared for self & cross) ----
  auto mha = [&](const u16* xq, const u16* xkv,
                 const u16* wqT, const float* bq, const u16* wkT, const float* bk,
                 const u16* wvT, const float* bv, const u16* woT, const float* bo,
                 const float* alphaP) {
    gemm(xq,  D, wqT, D, nullptr, 0, q_bf, D, bq, nullptr, nullptr, 0,
         M4, D, D, 1, 1, 1, 0, 0, 0, 0, 0, 0);
    gemm(xkv, D, wkT, D, nullptr, 0, k_bf, D, bk, nullptr, nullptr, 0,
         M4, D, D, 1, 1, 1, 0, 0, 0, 0, 0, 0);
    gemm(xkv, D, wvT, D, nullptr, 0, v_bf, D, bv, nullptr, nullptr, 0,
         M4, D, D, 1, 1, 1, 0, 0, 0, 0, 0, 0);
    // V -> V^T per head: [L,HD] -> [HD,L]
    k_tr_bf16<<<dim3(HD / 32, L / 32, Bz * H), dim3(32, 8), 0, stream>>>(
        v_bf, vt_bf, L, HD, D, H, (long long)L * D, (long long)HD,
        (long long)HD * L);
    // scores = Q @ K^T  (batched over b,h)
    gemm(q_bf, D, k_bf, D, scores, L, nullptr, 0, nullptr, nullptr, nullptr, 0,
         L, L, HD, 0, Bz * H, H,
         (long long)L * D, HD, (long long)L * D, HD,
         (long long)H * L * L, (long long)L * L);
    k_softmax<<<Bz * H * L, 256, 0, stream>>>(scores, aprobs, L, 1.0f / 32.0f);
    // ctx = probs @ V
    gemm(aprobs, L, vt_bf, L, nullptr, 0, ctx_bf, D, nullptr, nullptr,
         nullptr, 0, L, HD, L, 0, Bz * H, H,
         (long long)H * L * L, (long long)L * L,
         (long long)H * HD * L, (long long)HD * L,
         (long long)L * D, HD);
    // y = relu(ctx @ wo + bo); x = x + alpha * y  (fused residual)
    gemm(ctx_bf, D, woT, D, x_f, D, x_bf, D, bo, alphaP, x_f, D,
         M4, D, D, 1, 1, 1, 0, 0, 0, 0, 0, 0);
  };

  mha(x_bf, x_bf,  saWqT, sa_bq, saWkT, sa_bk, saWvT, sa_bv, saWoT, sa_bo, a_sa);
  mha(x_bf, src_bf, caWqT, ca_bq, caWkT, ca_bk, caWvT, ca_bv, caWoT, ca_bo, a_ca);

  // ---- feed-forward: leaky_relu(x@W1+b1)@W2+b2 ; out = x + alpha*ff ----
  gemm(x_bf, D, ff1T, D, nullptr, 0, hid, DFF, ff_b1, nullptr, nullptr, 0,
       M4, DFF, D, 2, 1, 1, 0, 0, 0, 0, 0, 0);
  gemm(hid, DFF, ff2T, DFF, (float*)d_out, D, nullptr, 0, ff_b2, a_ff, x_f, D,
       M4, D, DFF, 0, 1, 1, 0, 0, 0, 0, 0, 0);
}
